// MSDeformAttnTransformerEncoderOnly_6451040879008
// MI455X (gfx1250) — matrix operand order repaired and checked by reading, compiled, not run on hardware
//
#include <hip/hip_runtime.h>

// ---------------------------------------------------------------------------
// MSDeformAttn encoder for MI455X (gfx1250): bf16 WMMA GEMM/conv pipeline.
// - 128x64 block tile GEMM, 32x32 wave tiles (4 wmma / k-step / wave)
// - bf16-A GEMM stages LDS tiles with global_load_async_to_lds_b128 (ASYNCcnt)
// - implicit-im2col WMMA 3x3 conv, 2 outch tiles per wave
// ---------------------------------------------------------------------------

typedef __attribute__((ext_vector_type(16))) __bf16 v16bf;
typedef __attribute__((ext_vector_type(8)))  __bf16 v8bf;
typedef __attribute__((ext_vector_type(8)))  float  v8f;

__device__ __forceinline__ __bf16 f2bf(float f) {
  union { float f; unsigned u; } a; a.f = f;
  unsigned r = a.u + 0x7FFFu + ((a.u >> 16) & 1u);   // round-to-nearest-even
  union { unsigned short s; __bf16 b; } o; o.s = (unsigned short)(r >> 16);
  return o.b;
}
__device__ __forceinline__ float bf2f(__bf16 b) {
  union { __bf16 b; unsigned short s; } i; i.b = b;
  union { unsigned u; float f; } o; o.u = ((unsigned)i.s) << 16;
  return o.f;
}
__device__ __forceinline__ float toF(float x)  { return x; }
__device__ __forceinline__ float toF(__bf16 x) { return bf2f(x); }

__device__ __forceinline__ v16bf zero16() {
  union { v16bf v; int i[8]; } u;
#pragma unroll
  for (int k = 0; k < 8; ++k) u.i[k] = 0;
  return u.v;
}

// A-fragment per CDNA5 ISA 16-bit A 16x32 layout:
// lane<16: M=lane, K={0..7,16..23}; lane>=16: M=lane-16, K={8..15,24..31}
__device__ __forceinline__ v16bf load_a_frag(const __bf16* rowbase, int stride, int lane) {
  const int l = lane & 15;
  const __bf16* p = rowbase + l * stride + ((lane >= 16) ? 8 : 0);
  v8bf c0 = *(const v8bf*)(p);
  v8bf c1 = *(const v8bf*)(p + 16);
  union { v16bf v; v8bf h[2]; } u; u.h[0] = c0; u.h[1] = c1;
  return u.v;
}

// ---------------------------------------------------------------------------
// Generic GEMM: D[M,N] = A[M,K] @ W^T (+bias) with epilogues.
// A: f32 or bf16 row-major; B: pre-packed bf16 fragments (512 el per 32x16 frag,
// frag index kt*NT+nt, within frag: lane*16+j).
// mode 0: outF = acc + bias
// mode 1: outF = acc + bias + resid
// mode 2: outH(bf16) = s*relu(acc)^2 + b   (star MLP hidden)
// Block = 256 thr = 8 waves (4 M x 2 N); block tile 128x64; wave tile 32x32.
// ---------------------------------------------------------------------------
template <typename AT>
__global__ __launch_bounds__(256) void k_gemm(
    const AT* __restrict__ A, const __bf16* __restrict__ Bp,
    const float* __restrict__ bias, const float* __restrict__ resid,
    float* __restrict__ outF, __bf16* __restrict__ outH,
    const float* __restrict__ starS, const float* __restrict__ starB, int lidx,
    int M, int N, int K, int NT, int mode)
{
  __shared__ __bf16 As[128 * 40];     // 128 rows x 32 cols, stride 40 (16B-aligned rows)
  const int t = threadIdx.x;
  const int lane = t & 31, wave = t >> 5;
  const int wm = wave >> 1;           // 0..3 -> 32-row strip
  const int wn = wave & 1;            // 0..1 -> 32-col strip
  const int m0 = blockIdx.x * 128, n0 = blockIdx.y * 64;
  const int ntA = (n0 >> 4) + wn * 2;
  const int ntB = ntA + 1;

  v8f acc00 = {}; v8f acc01 = {};     // acc[mtile][ntile]
  v8f acc10 = {}; v8f acc11 = {};
  const int KT = K >> 5;
  for (int kt = 0; kt < KT; ++kt) {
    const int k0 = kt << 5;
    if constexpr (sizeof(AT) == 2) {
      // A already bf16: gfx1250 async DMA global -> LDS, 16B chunks (ASYNCcnt).
#pragma unroll
      for (int i = 0; i < 2; ++i) {
        int chunk = t * 2 + i;                 // 512 x 16B chunks = 128 rows x 64B
        int r = chunk >> 2, sub = chunk & 3;
        int gm = m0 + r;
        if (gm < M) {
          unsigned ldsoff = (unsigned)(r * 80 + sub * 16);   // As is at LDS offset 0
          const char* g = (const char*)(A + (size_t)gm * K + k0) + sub * 16;
          asm volatile("global_load_async_to_lds_b128 %0, %1, off"
                       :: "v"(ldsoff), "v"(g) : "memory");
        }
      }
      asm volatile("s_wait_asynccnt 0x0" ::: "memory");
    } else {
#pragma unroll
      for (int i = 0; i < 16; ++i) {           // stage 128x32 A tile, f32->bf16
        int idx = t * 16 + i;
        int r = idx >> 5, c = idx & 31;
        int gm = m0 + r;
        float v = (gm < M) ? toF(A[(size_t)gm * K + (k0 + c)]) : 0.f;
        As[r * 40 + c] = f2bf(v);
      }
    }
    __syncthreads();

    v16bf b0 = (ntA * 16 < N)
        ? *(const v16bf*)(Bp + ((size_t)(kt * NT + ntA) * 512 + lane * 16)) : zero16();
    v16bf b1 = (ntB * 16 < N)
        ? *(const v16bf*)(Bp + ((size_t)(kt * NT + ntB) * 512 + lane * 16)) : zero16();
    v16bf a0 = load_a_frag(&As[(wm * 32) * 40],      40, lane);
    v16bf a1 = load_a_frag(&As[(wm * 32 + 16) * 40], 40, lane);
    acc00 = __builtin_amdgcn_wmma_f32_16x16x32_bf16(false, a0, false, b0, (short)0, acc00, false, false);
    acc01 = __builtin_amdgcn_wmma_f32_16x16x32_bf16(false, a0, false, b1, (short)0, acc01, false, false);
    acc10 = __builtin_amdgcn_wmma_f32_16x16x32_bf16(false, a1, false, b0, (short)0, acc10, false, false);
    acc11 = __builtin_amdgcn_wmma_f32_16x16x32_bf16(false, a1, false, b1, (short)0, acc11, false, false);

    if (k0 + 32 < K) {                // gfx1250 global_prefetch_b8 of next K tile
      int gm = m0 + (t & 127);
      if (gm < M) __builtin_prefetch(&A[(size_t)gm * K + (k0 + 32)], 0, 1);
    }
    __syncthreads();
  }

  // D layout: lane<16 -> N=lane, M=r; lane>=16 -> N=lane-16, M=r+8
  const int nl = lane & 15;
  const int mo = (lane < 16) ? 0 : 8;
  float ss = 0.f, sb = 0.f;
  if (mode == 2) { ss = starS[lidx]; sb = starB[lidx]; }
#pragma unroll
  for (int nti = 0; nti < 2; ++nti) {
    int n = n0 + wn * 32 + nti * 16 + nl;
    if (n >= N) continue;
    float bv = (bias != nullptr) ? bias[n] : 0.f;
#pragma unroll
    for (int r = 0; r < 8; ++r) {
      int mg0 = m0 + wm * 32 + r + mo;
      int mg1 = mg0 + 16;
      float v0 = (nti ? acc01[r] : acc00[r]) + bv;
      float v1 = (nti ? acc11[r] : acc10[r]) + bv;
      if (mode == 0) {
        if (mg0 < M) outF[(size_t)mg0 * N + n] = v0;
        if (mg1 < M) outF[(size_t)mg1 * N + n] = v1;
      } else if (mode == 1) {
        if (mg0 < M) outF[(size_t)mg0 * N + n] = v0 + resid[(size_t)mg0 * N + n];
        if (mg1 < M) outF[(size_t)mg1 * N + n] = v1 + resid[(size_t)mg1 * N + n];
      } else {
        if (mg0 < M) { float rl = v0 > 0.f ? v0 : 0.f; outH[(size_t)mg0 * N + n] = f2bf(ss * rl * rl + sb); }
        if (mg1 < M) { float rl = v1 > 0.f ? v1 : 0.f; outH[(size_t)mg1 * N + n] = f2bf(ss * rl * rl + sb); }
      }
    }
  }
}

// ---------------------------------------------------------------------------
// 3x3 'SAME' conv, NHWC, C=256->256, implicit-im2col WMMA.
// Block: 16-pixel row segment; LDS stages 3 rows x 18 cols x 256ch (bf16, 27KB).
// 8 waves x 2 outch tiles cover all 256 output channels in one block.
// ---------------------------------------------------------------------------
__global__ __launch_bounds__(256) void k_conv3x3(
    const float* __restrict__ X, const __bf16* __restrict__ Wp,
    const float* __restrict__ bias, float* __restrict__ Y,
    int Bn, int H, int W)
{
  __shared__ __bf16 Xs[3 * 18 * 256];
  const int t = threadIdx.x, lane = t & 31, wave = t >> 5;
  const int tilesPerRow = W >> 4;
  int tmp = blockIdx.x;
  const int xt = tmp % tilesPerRow; tmp /= tilesPerRow;
  const int y = tmp % H; const int b = tmp / H;
  const int x0 = xt << 4;

  for (int i = t; i < 3 * 18 * 256; i += 256) {
    int c = i & 255; int col = (i >> 8) % 18; int r = i / (18 * 256);
    int yy = y + r - 1, xx = x0 + col - 1;
    float v = 0.f;
    if (yy >= 0 && yy < H && xx >= 0 && xx < W)
      v = X[(((size_t)b * H + yy) * W + xx) * 256 + c];
    Xs[i] = f2bf(v);
  }
  __syncthreads();

  const int ntb = wave * 2;           // two 16-outch tiles per wave
  v8f acc0 = {}; v8f acc1 = {};
  for (int kh = 0; kh < 3; ++kh)
    for (int kw = 0; kw < 3; ++kw) {
#pragma unroll
      for (int kc = 0; kc < 8; ++kc) {
        v16bf af = load_a_frag(&Xs[(kh * 18 + kw) * 256 + kc * 32], 256, lane);
        const __bf16* wb = Wp + (size_t)((kh * 3 + kw) * 8 + kc) * 16 * 512 + lane * 16;
        v16bf bf0 = *(const v16bf*)(wb + (size_t)ntb * 512);
        v16bf bf1 = *(const v16bf*)(wb + (size_t)(ntb + 1) * 512);
        acc0 = __builtin_amdgcn_wmma_f32_16x16x32_bf16(false, af, false, bf0, (short)0, acc0, false, false);
        acc1 = __builtin_amdgcn_wmma_f32_16x16x32_bf16(false, af, false, bf1, (short)0, acc1, false, false);
      }
    }

  const int nl = lane & 15;
  const int na = ntb * 16 + nl, nb = na + 16;
  const int mo = (lane < 16) ? 0 : 8;
  const float bva = bias[na], bvb = bias[nb];
#pragma unroll
  for (int r = 0; r < 8; ++r) {
    int px = r + mo;
    size_t o = (((size_t)b * H + y) * W + (x0 + px)) * 256;
    Y[o + na] = acc0[r] + bva;
    Y[o + nb] = acc1[r] + bvb;
  }
}

// ---------------------------------------------------------------------------
// LayerNorm (C=256), one wave per token; optional second output with +pos.
// ---------------------------------------------------------------------------
__global__ __launch_bounds__(256) void k_ln(
    const float* __restrict__ x, const float* __restrict__ w,
    const float* __restrict__ pos, int posPeriod,
    float* __restrict__ out0, float* __restrict__ out1, int ntok)
{
  const int wave = threadIdx.x >> 5, lane = threadIdx.x & 31;
  const int tok = blockIdx.x * 8 + wave;
  if (tok >= ntok) return;
  const float* xr = x + (size_t)tok * 256;
  float v[8]; float s = 0.f, s2 = 0.f;
#pragma unroll
  for (int i = 0; i < 8; ++i) { v[i] = xr[lane + i * 32]; s += v[i]; s2 += v[i] * v[i]; }
#pragma unroll
  for (int o = 16; o > 0; o >>= 1) { s += __shfl_xor(s, o, 32); s2 += __shfl_xor(s2, o, 32); }
  const float m = s * (1.f / 256.f);
  const float var = s2 * (1.f / 256.f) - m * m;
  const float rinv = rsqrtf(var + 1e-6f);
  const float* pr = pos ? pos + (size_t)(tok % posPeriod) * 256 : nullptr;
#pragma unroll
  for (int i = 0; i < 8; ++i) {
    int c = lane + i * 32;
    float yv = (v[i] - m) * rinv * w[c];
    out0[(size_t)tok * 256 + c] = yv;
    if (out1) out1[(size_t)tok * 256 + c] = yv + pr[c];
  }
}

// ---------------------------------------------------------------------------
// Deformable attention sampling: one wave per (b, q, head); lane = channel.
// ---------------------------------------------------------------------------
__global__ __launch_bounds__(256) void k_msda(
    const float* __restrict__ value, const float* __restrict__ off,
    const float* __restrict__ aw, const float* __restrict__ ref,
    float* __restrict__ out)
{
  const int wave = threadIdx.x >> 5, lane = threadIdx.x & 31;
  const long gid = (long)blockIdx.x * 8 + wave;
  const int h = (int)(gid & 7);
  const long bq = gid >> 3;             // b*5424 + q
  const int q = (int)(bq % 5424);
  const int b = (int)(bq / 5424);
  if (b >= 4) return;

  const float* offp = off + bq * 192 + h * 24;
  const float* awp  = aw  + bq * 96  + h * 12;
  float a[12]; float mx = -1e30f;
#pragma unroll
  for (int i = 0; i < 12; ++i) { a[i] = awp[i]; mx = fmaxf(mx, a[i]); }
  float s = 0.f;
#pragma unroll
  for (int i = 0; i < 12; ++i) { a[i] = __expf(a[i] - mx); s += a[i]; }
  const float inv = 1.f / s;

  const int Hs[3] = {64, 32, 16};
  const int LSi[3] = {0, 4096, 5120};
  float o = 0.f;
#pragma unroll
  for (int l = 0; l < 3; ++l) {
    const int Hc = Hs[l], Wc = Hs[l];
    const float rx = ref[q * 6 + l * 2 + 0], ry = ref[q * 6 + l * 2 + 1];
    const float* vb = value + ((size_t)b * 5424 + 48 + LSi[l]) * 256 + h * 32 + lane;
#pragma unroll
    for (int p = 0; p < 4; ++p) {
      const float ox = offp[l * 8 + p * 2 + 0], oy = offp[l * 8 + p * 2 + 1];
      const float xf = (rx + ox / (float)Wc) * Wc - 0.5f;
      const float yf = (ry + oy / (float)Hc) * Hc - 0.5f;
      const float x0f = floorf(xf), y0f = floorf(yf);
      const int x0 = (int)x0f, y0 = (int)y0f;
      const float wx = xf - x0f, wy = yf - y0f;
      const float wgt = a[l * 4 + p] * inv;
#pragma unroll
      for (int d = 0; d < 4; ++d) {
        const int dx = d & 1, dy = d >> 1;
        const int xi = x0 + dx, yi = y0 + dy;
        if (xi >= 0 && xi < Wc && yi >= 0 && yi < Hc) {
          const float tw = (dx ? wx : 1.f - wx) * (dy ? wy : 1.f - wy);
          o += wgt * tw * vb[(size_t)(yi * Wc + xi) * 256];
        }
      }
    }
  }
  out[bq * 256 + h * 32 + lane] = o;
}

// ---------------------------------------------------------------------------
// Bilinear upsample (half-pixel centers, edge clamp) + add map -> dst.
// ---------------------------------------------------------------------------
__global__ void k_upsample_add(
    const float* __restrict__ cur, size_t curBS, int Hsi, int Wsi,
    const float* __restrict__ map, size_t mapBS,
    float* __restrict__ dst, int Hd, int Wd, int Bn)
{
  size_t idx = (size_t)blockIdx.x * 256 + threadIdx.x;
  size_t total = (size_t)Bn * Hd * Wd * 256;
  if (idx >= total) return;
  int c = (int)(idx & 255);
  size_t r = idx >> 8;
  int x = (int)(r % Wd); r /= Wd;
  int y = (int)(r % Hd); int b = (int)(r / Hd);
  float sy = (y + 0.5f) * ((float)Hsi / Hd) - 0.5f;
  float sx = (x + 0.5f) * ((float)Wsi / Wd) - 0.5f;
  sy = fminf(fmaxf(sy, 0.f), (float)Hsi - 1.f);
  sx = fminf(fmaxf(sx, 0.f), (float)Wsi - 1.f);
  int y0 = (int)sy, x0 = (int)sx;
  int y1 = y0 + 1 < Hsi ? y0 + 1 : Hsi - 1;
  int x1 = x0 + 1 < Wsi ? x0 + 1 : Wsi - 1;
  float wy = sy - y0, wx = sx - x0;
  const float* cb = cur + (size_t)b * curBS;
  float v00 = cb[((size_t)y0 * Wsi + x0) * 256 + c];
  float v01 = cb[((size_t)y0 * Wsi + x1) * 256 + c];
  float v10 = cb[((size_t)y1 * Wsi + x0) * 256 + c];
  float v11 = cb[((size_t)y1 * Wsi + x1) * 256 + c];
  float v = v00 * (1 - wx) * (1 - wy) + v01 * wx * (1 - wy)
          + v10 * (1 - wx) * wy + v11 * wx * wy;
  dst[idx] = map[(size_t)b * mapBS + ((size_t)y * Wd + x) * 256 + c] + v;
}

__global__ void k_add(float* __restrict__ d, const float* __restrict__ a, size_t n) {
  size_t i = (size_t)blockIdx.x * 256 + threadIdx.x;
  if (i < n) d[i] += a[i];
}

// Sine positional embedding for all hw tokens (first 48 reg tokens = 0).
__global__ void k_sinepos(float* __restrict__ pos) {   // [5424,256]
  int idx = blockIdx.x * 256 + threadIdx.x;
  if (idx >= 5424 * 256) return;
  int tok = idx >> 8, c = idx & 255;
  float out = 0.f;
  if (tok >= 48) {
    int t2 = tok - 48; int H, W, base;
    if (t2 < 4096)      { H = 64; W = 64; base = 0; }
    else if (t2 < 5120) { H = 32; W = 32; base = 4096; }
    else                { H = 16; W = 16; base = 5120; }
    int i = t2 - base; int yy = i / W, xx = i % W;
    const float scale = 6.283185307179586f;
    float v = (c < 128) ? (yy + 1) / ((float)H + 1e-6f) * scale
                        : (xx + 1) / ((float)W + 1e-6f) * scale;
    int j = c & 127;
    float dim_t = __powf(10000.f, (float)(j & ~1) / 128.f);
    float p = v / dim_t;
    out = (j & 1) ? __cosf(p) : __sinf(p);
  }
  pos[idx] = out;
}

// Reference points [5424][3][2]; same (x,y) broadcast over the 3 level slots.
__global__ void k_refs(float* __restrict__ ref) {
  int idx = blockIdx.x * 256 + threadIdx.x;
  if (idx >= 5424 * 6) return;
  int tok = idx / 6; int xy = idx & 1;
  float v = 0.f;
  if (tok >= 48) {
    int t2 = tok - 48; int H, W, base;
    if (t2 < 4096)      { H = 64; W = 64; base = 0; }
    else if (t2 < 5120) { H = 32; W = 32; base = 4096; }
    else                { H = 16; W = 16; base = 5120; }
    int i = t2 - base; int yy = i / W, xx = i % W;
    v = (xy == 0) ? (xx + 0.5f) / W : (yy + 0.5f) / H;
  }
  ref[idx] = v;
}

// Pack linear weight f32 [L][N][K] -> bf16 WMMA B-fragments.
__global__ void k_pack_lin(const float* __restrict__ Wf, __bf16* __restrict__ dst,
                           int N, int K, int L)
{
  const int KT = K >> 5, NT = (N + 15) >> 4;
  const size_t per = (size_t)KT * NT * 512;
  const size_t total = per * L;
  size_t idx = (size_t)blockIdx.x * 256 + threadIdx.x;
  if (idx >= total) return;
  int e = (int)(idx % 512); size_t f = idx / 512;
  int fl = (int)(f % ((size_t)KT * NT)); int layer = (int)(f / ((size_t)KT * NT));
  int kt = fl / NT, nt = fl % NT;
  int lane = e >> 4, j = e & 15;
  int n = nt * 16 + (lane & 15);
  int k = kt * 32 + ((lane >= 16) ? 8 : 0) + (j & 7) + ((j >> 3) << 4);
  float v = (n < N) ? Wf[(size_t)layer * N * K + (size_t)n * K + k] : 0.f;
  dst[idx] = f2bf(v);
}

// Pack conv OIHW f32 [nmats][256][256][3][3] -> per-tap bf16 B-fragments.
__global__ void k_pack_conv(const float* __restrict__ Wf, __bf16* __restrict__ dst, int nmats)
{
  const size_t per = 9 * 8 * 16 * 512;   // 589824
  const size_t total = per * nmats;
  size_t idx = (size_t)blockIdx.x * 256 + threadIdx.x;
  if (idx >= total) return;
  int e = (int)(idx % 512); size_t f = idx / 512;
  int nt = (int)(f % 16); f /= 16;
  int kc = (int)(f % 8);  f /= 8;
  int tap = (int)(f % 9); int mat = (int)(f / 9);
  int lane = e >> 4, j = e & 15;
  int o = nt * 16 + (lane & 15);
  int ic = kc * 32 + ((lane >= 16) ? 8 : 0) + (j & 7) + ((j >> 3) << 4);
  int kh = tap / 3, kw = tap % 3;
  float v = Wf[(((size_t)mat * 256 + o) * 256 + ic) * 9 + kh * 3 + kw];
  dst[idx] = f2bf(v);
}

// ---------------------------------------------------------------------------
extern "C" void kernel_launch(void* const* d_in, const int* in_sizes, int n_in,
                              void* d_out, int out_size, void* d_ws, size_t ws_size,
                              hipStream_t stream)
{
  (void)in_sizes; (void)n_in; (void)out_size; (void)ws_size;
  const float* res2 = (const float*)d_in[0];
  const float* res3 = (const float*)d_in[1];
  const float* res4 = (const float*)d_in[2];
  const float* res5 = (const float*)d_in[3];
  const float* reg3 = (const float*)d_in[4];
  const float* reg4 = (const float*)d_in[5];
  const float* reg5 = (const float*)d_in[6];
  const float* norm1_w = (const float*)d_in[7];
  const float* norm2_w = (const float*)d_in[8];
  const float* off_w = (const float*)d_in[9];
  const float* off_b = (const float*)d_in[10];
  const float* aw_w  = (const float*)d_in[11];
  const float* aw_b  = (const float*)d_in[12];
  const float* val_w = (const float*)d_in[13];
  const float* val_b = (const float*)d_in[14];
  const float* out_w = (const float*)d_in[15];
  const float* out_b = (const float*)d_in[16];
  const float* conv_w = (const float*)d_in[17];
  const float* conv_b = (const float*)d_in[18];
  const float* fc1_w = (const float*)d_in[19];
  const float* fc2_w = (const float*)d_in[20];
  const float* star_s = (const float*)d_in[21];
  const float* star_b = (const float*)d_in[22];

  const int Bn = 4, C = 256, Q = 5424;
  const int NTOK = Bn * Q;          // 21696
  const int FT = Bn * 16384;        // 65536 fpn tokens
  const size_t QC = (size_t)Q * C;  // src per-batch stride

  char* wsp = (char*)d_ws;
  size_t curoff = 0;
  auto alloc = [&](size_t bytes) -> void* {
    void* p = wsp + curoff;
    curoff += (bytes + 255) & ~(size_t)255;
    return p;
  };

  float* src    = (float*)alloc((size_t)NTOK * C * 4);
  float* fpn    = (float*)alloc((size_t)FT * C * 4);
  float* pos    = (float*)alloc((size_t)Q * C * 4);
  float* ref    = (float*)alloc((size_t)Q * 6 * 4);
  float* src2n  = (float*)alloc((size_t)NTOK * C * 4);   // LN(src) (value input)
  float* src2p  = (float*)alloc((size_t)NTOK * C * 4);   // LN(src)+pos (query)
  float* value  = (float*)alloc((size_t)NTOK * C * 4);
  float* offb   = (float*)alloc((size_t)NTOK * 192 * 4);
  float* awbuf  = (float*)alloc((size_t)NTOK * 96 * 4);
  float* attn   = (float*)alloc((size_t)NTOK * C * 4);
  float* src2   = (float*)alloc((size_t)NTOK * C * 4);   // msda projected output
  float* fpn2   = (float*)alloc((size_t)FT * C * 4);
  float* tbuf   = (float*)alloc((size_t)FT * C * 4);     // LN scratch (max rows)
  __bf16* hbuf  = (__bf16*)alloc((size_t)FT * 1024 * 2); // star-MLP hidden (bf16)
  float* f32b   = (float*)alloc((size_t)Bn * 32 * 32 * C * 4);
  float* f64b   = (float*)alloc((size_t)Bn * 64 * 64 * C * 4);
  float* f128b  = (float*)alloc((size_t)FT * C * 4);
  float* cur32  = (float*)alloc((size_t)Bn * 32 * 32 * C * 4);
  float* cur64  = (float*)alloc((size_t)Bn * 64 * 64 * C * 4);
  float* cur128 = (float*)alloc((size_t)FT * C * 4);
  __bf16* valP  = (__bf16*)alloc((size_t)6 * 65536 * 2);
  __bf16* outP  = (__bf16*)alloc((size_t)6 * 65536 * 2);
  __bf16* offP  = (__bf16*)alloc((size_t)6 * 49152 * 2);
  __bf16* awP   = (__bf16*)alloc((size_t)6 * 24576 * 2);
  __bf16* fc1P  = (__bf16*)alloc((size_t)6 * 262144 * 2);
  __bf16* fc2P  = (__bf16*)alloc((size_t)6 * 262144 * 2);
  __bf16* convP = (__bf16*)alloc((size_t)18 * 589824 * 2);

  // ---- weight packing (once per launch, deterministic) ----
  auto packLin = [&](const float* Wf, __bf16* dst, int N, int K) {
    size_t total = (size_t)(K / 32) * ((N + 15) / 16) * 512 * 6;
    k_pack_lin<<<dim3((unsigned)((total + 255) / 256)), 256, 0, stream>>>(Wf, dst, N, K, 6);
  };
  packLin(val_w, valP, 256, 256);
  packLin(out_w, outP, 256, 256);
  packLin(off_w, offP, 192, 256);
  packLin(aw_w,  awP,  96,  256);
  packLin(fc1_w, fc1P, 1024, 256);
  packLin(fc2_w, fc2P, 256, 1024);
  {
    size_t total = (size_t)18 * 589824;
    k_pack_conv<<<dim3((unsigned)((total + 255) / 256)), 256, 0, stream>>>(conv_w, convP, 18);
  }

  // ---- positional embeddings / reference points ----
  k_sinepos<<<dim3((Q * C + 255) / 256), 256, 0, stream>>>(pos);
  k_refs<<<dim3((Q * 6 + 255) / 256), 256, 0, stream>>>(ref);

  // ---- assemble src = [reg3|reg4|reg5|res3|res4|res5] per batch; fpn = res2 ----
  for (int b = 0; b < Bn; ++b) {
    float* sb = src + (size_t)b * QC;
    hipMemcpyAsync(sb,            reg3 + (size_t)b * 16 * C,   (size_t)16 * C * 4,   hipMemcpyDeviceToDevice, stream);
    hipMemcpyAsync(sb + 16 * C,   reg4 + (size_t)b * 16 * C,   (size_t)16 * C * 4,   hipMemcpyDeviceToDevice, stream);
    hipMemcpyAsync(sb + 32 * C,   reg5 + (size_t)b * 16 * C,   (size_t)16 * C * 4,   hipMemcpyDeviceToDevice, stream);
    hipMemcpyAsync(sb + 48 * C,   res3 + (size_t)b * 4096 * C, (size_t)4096 * C * 4, hipMemcpyDeviceToDevice, stream);
    hipMemcpyAsync(sb + (48 + 4096) * C, res4 + (size_t)b * 1024 * C, (size_t)1024 * C * 4, hipMemcpyDeviceToDevice, stream);
    hipMemcpyAsync(sb + (48 + 5120) * C, res5 + (size_t)b * 256 * C,  (size_t)256 * C * 4,  hipMemcpyDeviceToDevice, stream);
  }
  hipMemcpyAsync(fpn, res2, (size_t)FT * C * 4, hipMemcpyDeviceToDevice, stream);

  auto gemmGrid = [](int M, int N) {
    return dim3((unsigned)((M + 127) / 128), (unsigned)((N + 63) / 64));
  };

  for (int l = 0; l < 6; ++l) {
    const float* n1 = norm1_w + l * C;
    const float* n2 = norm2_w + l * C;

    // LN(src) -> src2n (value input) and src2p (query = +pos)
    k_ln<<<dim3((NTOK + 7) / 8), 256, 0, stream>>>(src, n1, pos, Q, src2n, src2p, NTOK);

    // value / offset / attention-weight projections (WMMA)
    k_gemm<float><<<gemmGrid(NTOK, 256), 256, 0, stream>>>(
        src2n, valP + (size_t)l * 65536, val_b + l * 256, nullptr,
        value, nullptr, nullptr, nullptr, 0, NTOK, 256, 256, 16, 0);
    k_gemm<float><<<gemmGrid(NTOK, 192), 256, 0, stream>>>(
        src2p, offP + (size_t)l * 49152, off_b + l * 192, nullptr,
        offb, nullptr, nullptr, nullptr, 0, NTOK, 192, 256, 12, 0);
    k_gemm<float><<<gemmGrid(NTOK, 96), 256, 0, stream>>>(
        src2p, awP + (size_t)l * 24576, aw_b + l * 96, nullptr,
        awbuf, nullptr, nullptr, nullptr, 0, NTOK, 96, 256, 6, 0);

    // deformable sampling (gather-bound; VALU/vmem)
    k_msda<<<dim3((unsigned)NTOK), 256, 0, stream>>>(value, offb, awbuf, ref, attn);

    // output projection -> src2 (kept separate; feeds FPN maps), then residual
    k_gemm<float><<<gemmGrid(NTOK, 256), 256, 0, stream>>>(
        attn, outP + (size_t)l * 65536, out_b + l * 256, nullptr,
        src2, nullptr, nullptr, nullptr, 0, NTOK, 256, 256, 16, 0);
    k_add<<<dim3((unsigned)(((size_t)NTOK * C + 255) / 256)), 256, 0, stream>>>(src, src2, (size_t)NTOK * C);

    // LN(fpn) -> fpn2
    k_ln<<<dim3((FT + 7) / 8), 256, 0, stream>>>(fpn, n1, nullptr, 1, fpn2, nullptr, FT);

    // ---- FPN: 16 -> 32 -> 64 -> 128 with WMMA convs ----
    const float* map16 = src2 + (size_t)(48 + 5120) * C;
    const float* map32 = src2 + (size_t)(48 + 4096) * C;
    const float* map64 = src2 + (size_t)48 * C;

    k_upsample_add<<<dim3((unsigned)(((size_t)Bn * 32 * 32 * C + 255) / 256)), 256, 0, stream>>>(
        map16, QC, 16, 16, map32, QC, f32b, 32, 32, Bn);
    k_conv3x3<<<dim3((unsigned)(Bn * 32 * 2)), 256, 0, stream>>>(
        f32b, convP + (size_t)(l * 3 + 0) * 589824, conv_b + (size_t)(l * 3 + 0) * 256, cur32, Bn, 32, 32);

    k_upsample_add<<<dim3((unsigned)(((size_t)Bn * 64 * 64 * C + 255) / 256)), 256, 0, stream>>>(
        cur32, (size_t)32 * 32 * C, 32, 32, map64, QC, f64b, 64, 64, Bn);
    k_conv3x3<<<dim3((unsigned)(Bn * 64 * 4)), 256, 0, stream>>>(
        f64b, convP + (size_t)(l * 3 + 1) * 589824, conv_b + (size_t)(l * 3 + 1) * 256, cur64, Bn, 64, 64);

    k_upsample_add<<<dim3((unsigned)(((size_t)FT * C + 255) / 256)), 256, 0, stream>>>(
        cur64, (size_t)64 * 64 * C, 64, 64, fpn2, (size_t)16384 * C, f128b, 128, 128, Bn);
    k_conv3x3<<<dim3((unsigned)(Bn * 128 * 8)), 256, 0, stream>>>(
        f128b, convP + (size_t)(l * 3 + 2) * 589824, conv_b + (size_t)(l * 3 + 2) * 256, cur128, Bn, 128, 128);

    k_add<<<dim3((unsigned)(((size_t)FT * C + 255) / 256)), 256, 0, stream>>>(fpn, cur128, (size_t)FT * C);

    // ---- star MLP on src: fc1 (star epilogue -> bf16 hidden), fc2 (+residual) ----
    k_ln<<<dim3((NTOK + 7) / 8), 256, 0, stream>>>(src, n2, nullptr, 1, tbuf, nullptr, NTOK);
    k_gemm<float><<<gemmGrid(NTOK, 1024), 256, 0, stream>>>(
        tbuf, fc1P + (size_t)l * 262144, nullptr, nullptr,
        nullptr, hbuf, star_s, star_b, l, NTOK, 1024, 256, 64, 2);
    k_gemm<__bf16><<<gemmGrid(NTOK, 256), 256, 0, stream>>>(
        hbuf, fc2P + (size_t)l * 262144, nullptr, src,
        src, nullptr, nullptr, nullptr, 0, NTOK, 256, 1024, 16, 1);

    // ---- star MLP on fpn ----
    k_ln<<<dim3((FT + 7) / 8), 256, 0, stream>>>(fpn, n2, nullptr, 1, tbuf, nullptr, FT);
    k_gemm<float><<<gemmGrid(FT, 1024), 256, 0, stream>>>(
        tbuf, fc1P + (size_t)l * 262144, nullptr, nullptr,
        nullptr, hbuf, star_s, star_b, l, FT, 1024, 256, 64, 2);
    k_gemm<__bf16><<<gemmGrid(FT, 256), 256, 0, stream>>>(
        hbuf, fc2P + (size_t)l * 262144, nullptr, fpn,
        fpn, nullptr, nullptr, nullptr, 0, FT, 256, 1024, 16, 1);
  }

  // ---- outputs: (src, fpn) concatenated flat ----
  hipMemcpyAsync((float*)d_out, src, (size_t)NTOK * C * 4, hipMemcpyDeviceToDevice, stream);
  hipMemcpyAsync((float*)d_out + (size_t)NTOK * C, fpn, (size_t)FT * C * 4, hipMemcpyDeviceToDevice, stream);
}
